// cait_models_1881195675989
// MI455X (gfx1250) — compile-verified, matrix operand order checked by reading, not run on hardware
//
#include <hip/hip_runtime.h>

// ---------------------------------------------------------------------------
// CaiT forward for MI455X (gfx1250, wave32, WMMA).
// All heavy matmuls go through one batched bf16 WMMA GEMM kernel
// (v_wmma_f32_16x16x32_bf16, fp32 accumulate) with double-buffered LDS fed by
// CDNA5 async global->LDS loads (GLOBAL_LOAD_ASYNC_TO_LDS_B128 / ASYNCcnt),
// plus fused bias/GELU/layer-scale-residual epilogues.
// ---------------------------------------------------------------------------

typedef unsigned short u16;
typedef __attribute__((ext_vector_type(16))) __bf16 v16bf;
typedef __attribute__((ext_vector_type(8)))  float  v8f;

__device__ __forceinline__ u16 f2b(float f) {        // f32 -> bf16 (RNE)
    unsigned u = __float_as_uint(f);
    u += 0x7FFFu + ((u >> 16) & 1u);
    return (u16)(u >> 16);
}
__device__ __forceinline__ float b2f(u16 h) {        // bf16 -> f32
    return __uint_as_float(((unsigned)h) << 16);
}

// low 32 bits of a flat shared-memory address == LDS byte offset
__device__ __forceinline__ unsigned lds_off(const void* p) {
    return (unsigned)(size_t)p;
}
// async copy of 16 bytes global -> LDS (per lane), tracked by ASYNCcnt
__device__ __forceinline__ void async_b128(unsigned lds, const unsigned* gsrc) {
    asm volatile("global_load_async_to_lds_b128 %0, %1, off"
                 :: "v"(lds), "v"(gsrc) : "memory");
}

// ------------------------------ WMMA GEMM ----------------------------------
// D[z][m][n] = epilogue( sum_k A[z][m][k] * W[z][n][k] )
// A: bf16 row-major [M x K], leading dim lda (elements, multiple of 8).
// W: bf16 row-major [N x K] (i.e. x @ W^T convention), leading dim ldw.
// z decomposes as (zb, zh) with zh in [0, innerB).
// mode: 0 = none, 1 = exact GELU, 2 = residual: D = Dold + gamma[n]*val.
#define GM_NONE 0
#define GM_GELU 1
#define GM_RES  2

#define LROW 20   // LDS row stride in dwords: 16 payload + 4 pad (16B aligned,
                  // and 20*r mod 64 distinct over r=0..15 -> conflict-free)

union AccU { v8f v; float f[8]; };
union FragU { unsigned u[8]; v16bf v; };

__global__ __launch_bounds__(256) void gemm_bf16(
    const u16* __restrict__ A, int lda, long long sAb, long long sAh,
    const u16* __restrict__ W, int ldw, long long sWb, long long sWh,
    void* __restrict__ Dv,     int ldd, long long sDb, long long sDh,
    const float* __restrict__ bias, const float* __restrict__ gamma,
    const unsigned* __restrict__ zblk,
    int M, int N, int K, int innerB, int mode, int outBf16)
{
    __shared__ unsigned As[2][128 * LROW];
    __shared__ unsigned Bs[2][64 * LROW];

    const int z  = blockIdx.z;
    const int zb = z / innerB, zh = z % innerB;
    const u16* Ab = A + zb * sAb + zh * sAh;
    const u16* Wb = W + zb * sWb + zh * sWh;
    const long long dOff = (long long)zb * sDb + (long long)zh * sDh;

    const int tid   = threadIdx.x;
    const int lane  = tid & 31;
    const int wid   = tid >> 5;
    const int waveM = wid & 3;          // 4 waves along M (32 rows each)
    const int waveN = wid >> 2;         // 2 waves along N (32 cols each)
    const int l16   = lane & 15;
    const int hh    = lane >> 4;        // lane half

    const int blockM = blockIdx.x * 128;
    const int blockN = blockIdx.y * 64;

    v8f acc[2][2];
#pragma unroll
    for (int i = 0; i < 2; i++)
#pragma unroll
        for (int j = 0; j < 2; j++)
#pragma unroll
            for (int r = 0; r < 8; r++) acc[i][j][r] = 0.0f;

    // staging assignment: A tile 128 rows x 32 bf16 (two B128 per lane),
    //                     B tile  64 rows x 32 bf16 (one B128 per lane)
    const int arow = tid >> 1, ah = tid & 1;
    const int brow = tid >> 2, bq = tid & 3;
    const int ldaw = lda >> 1, ldww = ldw >> 1;

    const int  grow   = blockM + arow;
    const int  gn     = blockN + brow;
    const bool aValid = grow < M;
    const bool bValid = gn < N;
    const unsigned* aBase =
        (const unsigned*)Ab + (size_t)(aValid ? grow : 0) * (size_t)ldaw + ah * 8;
    const unsigned* bBase =
        (const unsigned*)Wb + (size_t)(bValid ? gn : 0) * (size_t)ldww + bq * 4;

    unsigned ldsA[2], ldsB[2];
#pragma unroll
    for (int b = 0; b < 2; b++) {
        ldsA[b] = lds_off(&As[b][arow * LROW + ah * 8]);
        ldsB[b] = lds_off(&Bs[b][brow * LROW + bq * 4]);
    }

    // exactly 3 async issues per wave per step (OOB lanes read a zero block,
    // keeping the ASYNCcnt accounting exact for s_wait_asynccnt)
    auto stage = [&](int buf, int k0) {
        const unsigned* sa = aValid ? (aBase + (k0 >> 1)) : zblk;
        async_b128(ldsA[buf],      sa);
        async_b128(ldsA[buf] + 16, aValid ? (aBase + (k0 >> 1) + 4) : zblk);
        async_b128(ldsB[buf],      bValid ? (bBase + (k0 >> 1)) : zblk);
    };

    const int nsteps = K >> 5;          // K is a multiple of 32
    stage(0, 0);                        // prologue prefetch

    for (int s = 0; s < nsteps; s++) {
        const int cur = s & 1;
        if (s + 1 < nsteps) {
            stage(cur ^ 1, (s + 1) << 5);       // prefetch next K-step
            asm volatile("s_wait_asynccnt 3" ::: "memory");  // cur buffer done
        } else {
            asm volatile("s_wait_asynccnt 0" ::: "memory");
        }
        __syncthreads();                // cur buffer visible to all waves

        // Fragment K-dword pattern (ISA 7.12.2, 16-bit A 16x32):
        // reg p<4 : dword 4*hh + p ; reg p>=4 : dword 8 + 4*hh + (p-4)
        FragU a[2], b[2];
#pragma unroll
        for (int i = 0; i < 2; i++) {
            const int r = waveM * 32 + i * 16 + l16;
#pragma unroll
            for (int p = 0; p < 8; p++) {
                const int dw = (p < 4) ? (4 * hh + p) : (8 + 4 * hh + (p - 4));
                a[i].u[p] = As[cur][r * LROW + dw];
            }
        }
#pragma unroll
        for (int j = 0; j < 2; j++) {
            const int c = waveN * 32 + j * 16 + l16;
#pragma unroll
            for (int p = 0; p < 8; p++) {
                const int dw = (p < 4) ? (4 * hh + p) : (8 + 4 * hh + (p - 4));
                b[j].u[p] = Bs[cur][c * LROW + dw];
            }
        }
#pragma unroll
        for (int i = 0; i < 2; i++)
#pragma unroll
            for (int j = 0; j < 2; j++)
                acc[i][j] = __builtin_amdgcn_wmma_f32_16x16x32_bf16(
                    false, a[i].v, false, b[j].v, (short)0, acc[i][j],
                    false, false);

        __syncthreads();   // all waves done reading cur before it is re-staged
    }

    // epilogue (C/D layout: VGPR r holds rows r (lanes 0-15) and r+8 (16-31))
    float* Df = (float*)Dv;
    u16*   Db = (u16*)Dv;
#pragma unroll
    for (int i = 0; i < 2; i++)
#pragma unroll
        for (int j = 0; j < 2; j++) {
            const int col = blockN + waveN * 32 + j * 16 + l16;
            if (col >= N) continue;
            const float bi = bias  ? bias[col]  : 0.0f;
            const float gm = (mode == GM_RES && gamma) ? gamma[col] : 0.0f;
            AccU t; t.v = acc[i][j];
#pragma unroll
            for (int r = 0; r < 8; r++) {
                const int row = blockM + waveM * 32 + i * 16 + hh * 8 + r;
                if (row >= M) continue;
                const long long idx = dOff + (long long)row * ldd + col;
                float v = t.f[r] + bi;
                if (mode == GM_GELU)      v = 0.5f * v * (1.0f + erff(v * 0.70710678f));
                else if (mode == GM_RES)  v = Df[idx] + gm * v;
                if (outBf16) Db[idx] = f2b(v);
                else         Df[idx] = v;
            }
        }
}

// ------------------------------ LayerNorm ----------------------------------
// One wave32 per token, C=384 (12 elems/lane). Writes bf16.
__global__ __launch_bounds__(32) void ln384(
    const float* __restrict__ x, const float* __restrict__ g,
    const float* __restrict__ b, u16* __restrict__ out, int ntok)
{
    const int tok = blockIdx.x;
    if (tok >= ntok) return;
    const int lane = threadIdx.x;
    const float* xp = x + (size_t)tok * 384;
    float vals[12];
    float s = 0.0f;
#pragma unroll
    for (int i = 0; i < 12; i++) { vals[i] = xp[lane + 32 * i]; s += vals[i]; }
#pragma unroll
    for (int o = 16; o > 0; o >>= 1) s += __shfl_xor(s, o, 32);
    const float mean = s * (1.0f / 384.0f);
    float v2 = 0.0f;
#pragma unroll
    for (int i = 0; i < 12; i++) { float d = vals[i] - mean; v2 += d * d; }
#pragma unroll
    for (int o = 16; o > 0; o >>= 1) v2 += __shfl_xor(v2, o, 32);
    const float rstd = rsqrtf(v2 * (1.0f / 384.0f) + 1e-5f);
#pragma unroll
    for (int i = 0; i < 12; i++) {
        const int c = lane + 32 * i;
        out[(size_t)tok * 384 + c] = f2b((vals[i] - mean) * rstd * g[c] + b[c]);
    }
}

// ------------------------- elementwise helpers -----------------------------
__global__ void cvt_bf16(const float* __restrict__ in, u16* __restrict__ out,
                         long long n) {
    long long i = (long long)blockIdx.x * blockDim.x + threadIdx.x;
    if (i < n) out[i] = f2b(in[i]);
}

__global__ void zero_u32(unsigned* __restrict__ p, int n) {
    int i = blockIdx.x * blockDim.x + threadIdx.x;
    if (i < n) p[i] = 0u;
}

// extract non-overlapping 16x16 patches into xp[B*196, 768] (bf16)
__global__ void patch_extract(const float* __restrict__ x, u16* __restrict__ xp) {
    long long idx = (long long)blockIdx.x * blockDim.x + threadIdx.x;
    if (idx >= 12544LL * 768) return;
    const int col = (int)(idx % 768);
    long long r = idx / 768;
    const int pn = (int)(r % 196);
    const int b  = (int)(r / 196);
    const int cin = col / 256, rr = col % 256;
    const int py = rr / 16, px = rr % 16;
    const int gy = pn / 14, gx = pn % 14;
    const float v = x[(((long long)b * 3 + cin) * 224 + gy * 16 + py) * 224 +
                      gx * 16 + px];
    xp[idx] = f2b(v);
}

__global__ void add_pos(float* __restrict__ h, const float* __restrict__ pos) {
    long long idx = (long long)blockIdx.x * blockDim.x + threadIdx.x;
    if (idx >= 12544LL * 384) return;
    const int c = (int)(idx % 384);
    const int n = (int)((idx / 384) % 196);
    h[idx] += pos[n * 384 + c];
}

// split qkv (bf16 [B*196,1152]) -> qs/ks bf16 [B,NH,196,64] (HD 48 padded to 64)
__global__ void stage_qk(const u16* __restrict__ qkv, u16* __restrict__ qs,
                         u16* __restrict__ ks) {
    long long idx = (long long)blockIdx.x * blockDim.x + threadIdx.x;
    if (idx >= 64LL * 8 * 196 * 64) return;
    const int d = (int)(idx & 63);
    long long r = idx >> 6;
    const int n = (int)(r % 196); r /= 196;
    const int h = (int)(r % 8);
    const int b = (int)(r / 8);
    u16 qv = 0, kv = 0;
    if (d < 48) {
        const long long base = ((long long)(b * 196 + n)) * 1152 + h * 48 + d;
        qv = f2b(b2f(qkv[base]) * 0.14433757f);  // 1/sqrt(48)
        kv = qkv[base + 384];
    }
    qs[idx] = qv; ks[idx] = kv;
}

// v transposed: vT[B,NH,48,224] (token dim 196 padded to 224 with zeros)
__global__ void stage_v(const u16* __restrict__ qkv, u16* __restrict__ vT) {
    long long idx = (long long)blockIdx.x * blockDim.x + threadIdx.x;
    if (idx >= 64LL * 8 * 48 * 224) return;
    const int m = (int)(idx % 224);
    long long r = idx / 224;
    const int d = (int)(r % 48); r /= 48;
    const int h = (int)(r % 8);
    const int b = (int)(r / 8);
    u16 val = 0;
    if (m < 196)
        val = qkv[((long long)(b * 196 + m)) * 1152 + 768 + h * 48 + d];
    vT[idx] = val;
}

// talking-heads: mix(plw)+bias -> softmax over m<196 -> mix(pww)+bias -> bf16
__global__ __launch_bounds__(256) void talking_softmax(
    const u16* __restrict__ logits,                 // bf16 [B,NH,196,224]
    const float* __restrict__ plw, const float* __restrict__ plb,
    const float* __restrict__ pww, const float* __restrict__ pwb,
    u16* __restrict__ out)                          // bf16 [B,NH,196,224]
{
    const int n = blockIdx.x;       // 0..195
    const int b = blockIdx.y;       // 0..63
    const int m = threadIdx.x;      // 0..255
    const bool act = m < 196;
    __shared__ float red[256];

    float L[8];
    if (act) {
#pragma unroll
        for (int h = 0; h < 8; h++)
            L[h] = b2f(logits[((long long)(b * 8 + h) * 196 + n) * 224 + m]);
    }
    float m1[8];
    if (act) {
#pragma unroll
        for (int g = 0; g < 8; g++) {
            float s = plb[g];
#pragma unroll
            for (int h = 0; h < 8; h++) s += L[h] * plw[g * 8 + h];
            m1[g] = s;
        }
    }
    float soft[8];
#pragma unroll
    for (int g = 0; g < 8; g++) {
        red[m] = act ? m1[g] : -1e30f;
        __syncthreads();
        for (int s = 128; s > 0; s >>= 1) {
            if (m < s) red[m] = fmaxf(red[m], red[m + s]);
            __syncthreads();
        }
        const float mx = red[0];
        __syncthreads();
        const float e = act ? expf(m1[g] - mx) : 0.0f;
        red[m] = e;
        __syncthreads();
        for (int s = 128; s > 0; s >>= 1) {
            if (m < s) red[m] += red[m + s];
            __syncthreads();
        }
        const float sm = red[0];
        __syncthreads();
        soft[g] = act ? e / sm : 0.0f;
    }
    if (m < 224) {
#pragma unroll
        for (int g = 0; g < 8; g++) {
            float v = 0.0f;
            if (act) {
                v = pwb[g];
#pragma unroll
                for (int h = 0; h < 8; h++) v += soft[h] * pww[g * 8 + h];
            }
            out[((long long)(b * 8 + g) * 196 + n) * 224 + m] = f2b(v);
        }
    }
}

__global__ void cls_init(const float* __restrict__ tok, float* __restrict__ cls) {
    int i = blockIdx.x * blockDim.x + threadIdx.x;
    if (i < 64 * 384) cls[i] = tok[i % 384];
}

// u[B,197,384] = concat(cls, h)
__global__ void copy_u(const float* __restrict__ cls, const float* __restrict__ h,
                       float* __restrict__ u) {
    long long idx = (long long)blockIdx.x * blockDim.x + threadIdx.x;
    if (idx >= 64LL * 197 * 384) return;
    const int c = (int)(idx % 384);
    long long r = idx / 384;
    const int t = (int)(r % 197);
    const int b = (int)(r / 197);
    u[idx] = (t == 0) ? cls[b * 384 + c]
                      : h[((long long)b * 196 + (t - 1)) * 384 + c];
}

// class attention, one block per (b, h): 197 keys, HD=48. Writes bf16 o.
__global__ __launch_bounds__(256) void ca_attn(
    const float* __restrict__ q,   // [B, 384]
    const float* __restrict__ k,   // [B, 197, 384]
    const float* __restrict__ v,   // [B, 197, 384]
    u16* __restrict__ o)           // bf16 [B, 384]
{
    const int b = blockIdx.x, h = blockIdx.y;
    const int t = threadIdx.x;
    __shared__ float prob[256];
    __shared__ float red[256];

    float lg = -1e30f;
    if (t < 197) {
        const float* kp = k + ((long long)(b * 197 + t)) * 384 + h * 48;
        const float* qp = q + (long long)b * 384 + h * 48;
        float s = 0.0f;
#pragma unroll
        for (int d = 0; d < 48; d++) s += qp[d] * kp[d];
        lg = s * 0.14433757f;
    }
    red[t] = lg; __syncthreads();
    for (int s = 128; s > 0; s >>= 1) {
        if (t < s) red[t] = fmaxf(red[t], red[t + s]);
        __syncthreads();
    }
    const float mx = red[0]; __syncthreads();
    const float e = (t < 197) ? expf(lg - mx) : 0.0f;
    red[t] = e; __syncthreads();
    for (int s = 128; s > 0; s >>= 1) {
        if (t < s) red[t] += red[t + s];
        __syncthreads();
    }
    const float sum = red[0]; __syncthreads();
    prob[t] = e / sum; __syncthreads();
    if (t < 48) {
        float s = 0.0f;
        for (int j = 0; j < 197; j++)
            s += prob[j] * v[((long long)(b * 197 + j)) * 384 + h * 48 + t];
        o[(long long)b * 384 + h * 48 + t] = f2b(s);
    }
}

// ------------------------------ host side ----------------------------------
extern "C" void kernel_launch(void* const* d_in, const int* in_sizes, int n_in,
                              void* d_out, int out_size, void* d_ws,
                              size_t ws_size, hipStream_t stream)
{
    const float* x        = (const float*)d_in[0];
    const float* patch_w  = (const float*)d_in[1];
    const float* patch_b  = (const float*)d_in[2];
    const float* cls_tok  = (const float*)d_in[3];
    const float* pos_emb  = (const float*)d_in[4];
    const float* ln1_g    = (const float*)d_in[5];
    const float* ln1_b    = (const float*)d_in[6];
    const float* qkv_w    = (const float*)d_in[7];
    const float* proj_w   = (const float*)d_in[8];
    const float* proj_b   = (const float*)d_in[9];
    const float* pl_w     = (const float*)d_in[10];
    const float* pl_b     = (const float*)d_in[11];
    const float* pw_w     = (const float*)d_in[12];
    const float* pw_b     = (const float*)d_in[13];
    const float* ln2_g    = (const float*)d_in[14];
    const float* ln2_b    = (const float*)d_in[15];
    const float* fc1_w    = (const float*)d_in[16];
    const float* fc1_b    = (const float*)d_in[17];
    const float* fc2_w    = (const float*)d_in[18];
    const float* fc2_b    = (const float*)d_in[19];
    const float* g1       = (const float*)d_in[20];
    const float* g2       = (const float*)d_in[21];
    const float* ca_ln1_g = (const float*)d_in[22];
    const float* ca_ln1_b = (const float*)d_in[23];
    const float* ca_q_w   = (const float*)d_in[24];
    const float* ca_k_w   = (const float*)d_in[25];
    const float* ca_v_w   = (const float*)d_in[26];
    const float* ca_pr_w  = (const float*)d_in[27];
    const float* ca_pr_b  = (const float*)d_in[28];
    const float* ca_ln2_g = (const float*)d_in[29];
    const float* ca_ln2_b = (const float*)d_in[30];
    const float* ca_f1_w  = (const float*)d_in[31];
    const float* ca_f1_b  = (const float*)d_in[32];
    const float* ca_f2_w  = (const float*)d_in[33];
    const float* ca_f2_b  = (const float*)d_in[34];
    const float* ca_g1    = (const float*)d_in[35];
    const float* ca_g2    = (const float*)d_in[36];
    const float* norm_g   = (const float*)d_in[37];
    const float* norm_b   = (const float*)d_in[38];
    const float* head_w   = (const float*)d_in[39];
    const float* head_b   = (const float*)d_in[40];

    char* ws = (char*)d_ws;
    size_t off = 0;
    auto A8 = [&](size_t bytes) {
        size_t o = off;
        off += (bytes + 255) & ~(size_t)255;
        return o;
    };

    // bf16 weight copies
    const size_t oWpatch = A8(2ull * 384 * 768);
    const size_t oWqkv   = A8(2ull * 12 * 1152 * 384);
    const size_t oWproj  = A8(2ull * 12 * 384 * 384);
    const size_t oWfc1   = A8(2ull * 12 * 1536 * 384);
    const size_t oWfc2   = A8(2ull * 12 * 384 * 1536);
    const size_t oWcaq   = A8(2ull * 2 * 384 * 384);
    const size_t oWcak   = A8(2ull * 2 * 384 * 384);
    const size_t oWcav   = A8(2ull * 2 * 384 * 384);
    const size_t oWcapr  = A8(2ull * 2 * 384 * 384);
    const size_t oWcaf1  = A8(2ull * 2 * 1536 * 384);
    const size_t oWcaf2  = A8(2ull * 2 * 384 * 1536);
    const size_t oWhead  = A8(2ull * 1000 * 384);
    const size_t oZero   = A8(256);                   // zero block for OOB asyncs
    // activations
    const size_t oH   = A8(4ull * 12544 * 384);       // residual stream (f32)
    const size_t oCls = A8(4ull * 64 * 384);          // cls stream (f32)
    const size_t oLn  = A8(2ull * 12544 * 384);       // LN output (bf16)
    const size_t oQkv = A8(2ull * 12544 * 1152);      // qkv out (bf16); aliases xp
    const size_t oHid = A8(2ull * 12544 * 1536);      // MLP hidden (bf16)
    // shared scratch region: SA-attention buffers / CA buffers (time-disjoint)
    size_t sa = 0;
    auto SA8 = [&](size_t b) { size_t o = sa; sa += (b + 255) & ~(size_t)255; return o; };
    const size_t oQs = SA8(2ull * 64 * 8 * 196 * 64);
    const size_t oKs = SA8(2ull * 64 * 8 * 196 * 64);
    const size_t oVT = SA8(2ull * 64 * 8 * 48 * 224);
    const size_t oLg = SA8(2ull * 64 * 8 * 196 * 224);
    const size_t oAt = SA8(2ull * 64 * 8 * 196 * 224);
    const size_t oOc = SA8(2ull * 12544 * 384);
    size_t ca = 0;
    auto CA8 = [&](size_t b) { size_t o = ca; ca += (b + 255) & ~(size_t)255; return o; };
    const size_t oU   = CA8(4ull * 64 * 197 * 384);
    const size_t oUn  = CA8(2ull * 64 * 197 * 384);
    const size_t oKb  = CA8(4ull * 64 * 197 * 384);
    const size_t oVb  = CA8(4ull * 64 * 197 * 384);
    const size_t oQc  = CA8(4ull * 64 * 384);
    const size_t oCao = CA8(2ull * 64 * 384);
    const size_t oCln = CA8(2ull * 64 * 384);
    const size_t oChd = CA8(2ull * 64 * 1536);
    const size_t oScr = A8(sa > ca ? sa : ca);
    (void)ws_size; (void)in_sizes; (void)n_in; (void)out_size;

    u16* wPatch = (u16*)(ws + oWpatch);
    u16* wQkv   = (u16*)(ws + oWqkv);
    u16* wProj  = (u16*)(ws + oWproj);
    u16* wFc1   = (u16*)(ws + oWfc1);
    u16* wFc2   = (u16*)(ws + oWfc2);
    u16* wCaq   = (u16*)(ws + oWcaq);
    u16* wCak   = (u16*)(ws + oWcak);
    u16* wCav   = (u16*)(ws + oWcav);
    u16* wCapr  = (u16*)(ws + oWcapr);
    u16* wCaf1  = (u16*)(ws + oWcaf1);
    u16* wCaf2  = (u16*)(ws + oWcaf2);
    u16* wHead  = (u16*)(ws + oWhead);
    unsigned* pZ = (unsigned*)(ws + oZero);
    float* pH   = (float*)(ws + oH);
    float* pCls = (float*)(ws + oCls);
    u16* pLn    = (u16*)(ws + oLn);
    u16* pQkv   = (u16*)(ws + oQkv);
    u16* pXp    = (u16*)(ws + oQkv);        // alias (dead before qkv is live)
    u16* pHid   = (u16*)(ws + oHid);
    u16* pQs    = (u16*)(ws + oScr + oQs);
    u16* pKs    = (u16*)(ws + oScr + oKs);
    u16* pVT    = (u16*)(ws + oScr + oVT);
    u16* pLg    = (u16*)(ws + oScr + oLg);
    u16* pAt    = (u16*)(ws + oScr + oAt);
    u16* pOc    = (u16*)(ws + oScr + oOc);
    float* pU   = (float*)(ws + oScr + oU);
    u16* pUn    = (u16*)(ws + oScr + oUn);
    float* pKb  = (float*)(ws + oScr + oKb);
    float* pVb  = (float*)(ws + oScr + oVb);
    float* pQc  = (float*)(ws + oScr + oQc);
    u16* pCao   = (u16*)(ws + oScr + oCao);
    u16* pCln   = (u16*)(ws + oScr + oCln);
    u16* pChd   = (u16*)(ws + oScr + oChd);

    auto cvt = [&](const float* src, u16* dst, long long n) {
        cvt_bf16<<<(unsigned)((n + 255) / 256), 256, 0, stream>>>(src, dst, n);
    };
    auto gemm = [&](const u16* A, int lda, long long sAb, long long sAh,
                    const u16* W, int ldw, long long sWb, long long sWh,
                    void* D, int ldd, long long sDb, long long sDh,
                    const float* bias, const float* gamma,
                    int M, int N, int K, int Z, int innerB, int mode, int obf) {
        dim3 grid((M + 127) / 128, (N + 63) / 64, Z);
        gemm_bf16<<<grid, 256, 0, stream>>>(A, lda, sAb, sAh, W, ldw, sWb, sWh,
                                            D, ldd, sDb, sDh, bias, gamma, pZ,
                                            M, N, K, innerB, mode, obf);
    };

    // -- init zero block + weight conversion (fp32 -> bf16, once per call) --
    zero_u32<<<1, 64, 0, stream>>>(pZ, 64);
    cvt(patch_w, wPatch, 384LL * 768);
    cvt(qkv_w,   wQkv,   12LL * 1152 * 384);
    cvt(proj_w,  wProj,  12LL * 384 * 384);
    cvt(fc1_w,   wFc1,   12LL * 1536 * 384);
    cvt(fc2_w,   wFc2,   12LL * 384 * 1536);
    cvt(ca_q_w,  wCaq,   2LL * 384 * 384);
    cvt(ca_k_w,  wCak,   2LL * 384 * 384);
    cvt(ca_v_w,  wCav,   2LL * 384 * 384);
    cvt(ca_pr_w, wCapr,  2LL * 384 * 384);
    cvt(ca_f1_w, wCaf1,  2LL * 1536 * 384);
    cvt(ca_f2_w, wCaf2,  2LL * 384 * 1536);
    cvt(head_w,  wHead,  1000LL * 384);

    // -- patch embed: xp[12544,768] @ patch_w^T + b, then + pos_embed --
    patch_extract<<<(unsigned)((12544LL * 768 + 255) / 256), 256, 0, stream>>>(x, pXp);
    gemm(pXp, 768, 0, 0, wPatch, 768, 0, 0, pH, 384, 0, 0,
         patch_b, nullptr, 12544, 384, 768, 1, 1, GM_NONE, 0);
    add_pos<<<(unsigned)((12544LL * 384 + 255) / 256), 256, 0, stream>>>(pH, pos_emb);

    // -- 12 self-attention blocks with talking heads --
    for (int l = 0; l < 12; l++) {
        ln384<<<12544, 32, 0, stream>>>(pH, ln1_g + l * 384, ln1_b + l * 384,
                                        pLn, 12544);
        // qkv (no bias), bf16 out
        gemm(pLn, 384, 0, 0, wQkv + (size_t)l * 1152 * 384, 384, 0, 0,
             pQkv, 1152, 0, 0, nullptr, nullptr,
             12544, 1152, 384, 1, 1, GM_NONE, 1);
        stage_qk<<<(unsigned)((64LL * 8 * 196 * 64 + 255) / 256), 256, 0, stream>>>(
            pQkv, pQs, pKs);
        stage_v<<<(unsigned)((64LL * 8 * 48 * 224 + 255) / 256), 256, 0, stream>>>(
            pQkv, pVT);
        // logits[b,h] = qs @ ks^T  (M=196, N=196, K=64; Z=512)
        gemm(pQs, 64, 8LL * 196 * 64, 196LL * 64,
             pKs, 64, 8LL * 196 * 64, 196LL * 64,
             pLg, 224, 8LL * 196 * 224, 196LL * 224,
             nullptr, nullptr, 196, 196, 64, 512, 8, GM_NONE, 1);
        talking_softmax<<<dim3(196, 64), 256, 0, stream>>>(
            pLg, pl_w + l * 64, pl_b + l * 8, pw_w + l * 64, pw_b + l * 8, pAt);
        // o[b,h] = attn @ v  (M=196, N=48, K=224; scattered into [B*196, 384])
        gemm(pAt, 224, 8LL * 196 * 224, 196LL * 224,
             pVT, 224, 8LL * 48 * 224, 48LL * 224,
             pOc, 384, 196LL * 384, 48,
             nullptr, nullptr, 196, 48, 224, 512, 8, GM_NONE, 1);
        // proj + layer-scale residual into h
        gemm(pOc, 384, 0, 0, wProj + (size_t)l * 384 * 384, 384, 0, 0,
             pH, 384, 0, 0, proj_b + l * 384, g1 + l * 384,
             12544, 384, 384, 1, 1, GM_RES, 0);
        // MLP
        ln384<<<12544, 32, 0, stream>>>(pH, ln2_g + l * 384, ln2_b + l * 384,
                                        pLn, 12544);
        gemm(pLn, 384, 0, 0, wFc1 + (size_t)l * 1536 * 384, 384, 0, 0,
             pHid, 1536, 0, 0, fc1_b + l * 1536, nullptr,
             12544, 1536, 384, 1, 1, GM_GELU, 1);
        gemm(pHid, 1536, 0, 0, wFc2 + (size_t)l * 384 * 1536, 1536, 0, 0,
             pH, 384, 0, 0, fc2_b + l * 384, g2 + l * 384,
             12544, 384, 1536, 1, 1, GM_RES, 0);
    }

    // -- 2 class-attention blocks --
    cls_init<<<(64 * 384 + 255) / 256, 256, 0, stream>>>(cls_tok, pCls);
    for (int l = 0; l < 2; l++) {
        copy_u<<<(unsigned)((64LL * 197 * 384 + 255) / 256), 256, 0, stream>>>(
            pCls, pH, pU);
        ln384<<<64 * 197, 32, 0, stream>>>(pU, ca_ln1_g + l * 384,
                                           ca_ln1_b + l * 384, pUn, 64 * 197);
        // q: only the 64 cls rows of un (lda = 197*384)
        gemm(pUn, 197 * 384, 0, 0, wCaq + (size_t)l * 384 * 384, 384, 0, 0,
             pQc, 384, 0, 0, nullptr, nullptr, 64, 384, 384, 1, 1, GM_NONE, 0);
        gemm(pUn, 384, 0, 0, wCak + (size_t)l * 384 * 384, 384, 0, 0,
             pKb, 384, 0, 0, nullptr, nullptr, 64 * 197, 384, 384, 1, 1, GM_NONE, 0);
        gemm(pUn, 384, 0, 0, wCav + (size_t)l * 384 * 384, 384, 0, 0,
             pVb, 384, 0, 0, nullptr, nullptr, 64 * 197, 384, 384, 1, 1, GM_NONE, 0);
        ca_attn<<<dim3(64, 8), 256, 0, stream>>>(pQc, pKb, pVb, pCao);
        gemm(pCao, 384, 0, 0, wCapr + (size_t)l * 384 * 384, 384, 0, 0,
             pCls, 384, 0, 0, ca_pr_b + l * 384, ca_g1 + l * 384,
             64, 384, 384, 1, 1, GM_RES, 0);
        ln384<<<64, 32, 0, stream>>>(pCls, ca_ln2_g + l * 384,
                                     ca_ln2_b + l * 384, pCln, 64);
        gemm(pCln, 384, 0, 0, wCaf1 + (size_t)l * 1536 * 384, 384, 0, 0,
             pChd, 1536, 0, 0, ca_f1_b + l * 1536, nullptr,
             64, 1536, 384, 1, 1, GM_GELU, 1);
        gemm(pChd, 1536, 0, 0, wCaf2 + (size_t)l * 384 * 1536, 1536, 0, 0,
             pCls, 384, 0, 0, ca_f2_b + l * 384, ca_g2 + l * 384,
             64, 384, 1536, 1, 1, GM_RES, 0);
    }

    // -- final norm (cls row only) + head --
    ln384<<<64, 32, 0, stream>>>(pCls, norm_g, norm_b, pCln, 64);
    gemm(pCln, 384, 0, 0, wHead, 384, 0, 0,
         (float*)d_out, 1000, 0, 0, head_b, nullptr,
         64, 1000, 384, 1, 1, GM_NONE, 0);
}
